// CrossModalAttention_76751065579702
// MI455X (gfx1250) — matrix operand order repaired and checked by reading, not compiled, not run on hardware
//
#include <hip/hip_runtime.h>
#include <cstdint>
#include <cstddef>

typedef __bf16 bf16;
typedef __bf16 v16bf __attribute__((ext_vector_type(16)));
typedef __bf16 v8bf  __attribute__((ext_vector_type(8)));
typedef float  v8f   __attribute__((ext_vector_type(8)));
typedef uint32_t u32x4 __attribute__((ext_vector_type(4)));
typedef uint32_t u32x8 __attribute__((ext_vector_type(8)));

#define DEV static __device__ __forceinline__

// ---------------- CDNA5 Tensor Data Mover (inline asm) --------------------
// One instruction DMAs a 2-D tile (rows x 32 bf16) global -> LDS, writing
// 64B of data + 16B pad per row (pad_interval=16 DWORDs, pad_amount=4 DWORDs)
// to match the 80-byte LDS row pitch. 2-D tensors need only D# groups 0-1.
DEV void tdm_load_tile_2d(uint32_t lds_byte_addr, const void* gaddr,
                          uint32_t row_stride_elems, uint32_t tile_rows) {
  const uint64_t ga = (uint64_t)(uintptr_t)gaddr;
  u32x4 g0;
  g0[0] = 1u;                                   // count=1, user descriptor
  g0[1] = lds_byte_addr;                        // lds_addr
  g0[2] = (uint32_t)ga;                         // global_addr[31:0]
  g0[3] = (uint32_t)(ga >> 32) | (2u << 30);    // global_addr[56:32] | type=2
  u32x8 g1;
  g1[0] = (1u << 16)                            // data_size = 2 bytes
        | (1u << 20)                            // pad_enable
        | (3u << 22)                            // pad_interval: 16 DWORDs
        | (3u << 25);                           // pad_amount:   4 DWORDs
  g1[1] = (row_stride_elems & 0xFFFFu) << 16;   // tensor_dim0[15:0]
  g1[2] = (row_stride_elems >> 16);             // tensor_dim0[31:16] | dim1 lo16(=0)
  g1[3] = (0x100000u >> 16)                     // tensor_dim1 = 1M rows (no OOB)
        | (32u << 16);                          // tile_dim0 = 32 elems
  g1[4] = tile_rows & 0xFFFFu;                  // tile_dim1 ; tile_dim2 = 0
  g1[5] = row_stride_elems;                     // tensor_dim0_stride[31:0]
  g1[6] = 0;                                    // stride0 hi | dim1_stride lo
  g1[7] = 0;                                    // dim1_stride hi
  asm volatile("tensor_load_to_lds %0, %1" :: "s"(g0), "s"(g1) : "memory");
}

// ---------------- f32 -> bf16 conversion ----------------------------------
__global__ __launch_bounds__(256)
void cvt_f32_to_bf16(const float* __restrict__ src, bf16* __restrict__ dst, int n4) {
  int i = blockIdx.x * blockDim.x + threadIdx.x;
  if (i >= n4) return;
  const float4 f = ((const float4*)src)[i];
  union { bf16 h[4]; short4 s4; } u;
  u.h[0] = (bf16)f.x; u.h[1] = (bf16)f.y; u.h[2] = (bf16)f.z; u.h[3] = (bf16)f.w;
  ((short4*)dst)[i] = u.s4;
}

// ---------------- generic bf16 WMMA GEMM ----------------------------------
// C[b] = (A[b] @ Bt[b]^T + bias) * scale
//   A : [M,K] bf16, row stride K        (batch stride sAb elements)
//   Bt: [N,K] bf16, row stride ldB      (batch stride sBb elements)
//   C : normal  -> [M,ldC] (bf16 or f32), TRANS_OUT -> bf16 [N,ldC], elem (m,n) at n*ldC+m
// block: 256 thr = 8 waves (2 M x 4 N); block tile 64(M) x 256(N);
// wave tile 32x64 (2x4 WMMA frags); K-step 32; double-buffered TDM tile fill:
// wave0 DMAs the A tile, wave1 the B tile; TENSORcnt gates the buffers.
template<bool TRANS_OUT, bool OUT_F32>
__global__ __launch_bounds__(256)
void gemm_bf16_wmma(const bf16* __restrict__ A, int64_t sAb,
                    const bf16* __restrict__ Bt, int64_t sBb, int ldB,
                    const float* __restrict__ bias,
                    void* __restrict__ C, int64_t sCb, int ldC,
                    int M, int N, int K, float scale)
{
  __shared__ __align__(16) bf16 As[2][64][40];     // 32 halves + 8 pad (80B rows)
  __shared__ __align__(16) bf16 Bs[2][256][40];

  const int tid  = threadIdx.x;
  const int lane = tid & 31;
  const int wave = tid >> 5;
  const int wm   = wave & 1;        // 0..1 (M dir)
  const int wn   = wave >> 1;       // 0..3 (N dir)
  const int lh   = lane >> 4;       // lane half (0/1)
  const int l16  = lane & 15;

  const int m0 = blockIdx.y * 64;
  const int n0 = blockIdx.x * 256;
  const int bz = blockIdx.z;

  const bf16* Ab = A  + (size_t)bz * sAb + (size_t)m0 * K;    // A tile row 0
  const bf16* Bb = Bt + (size_t)bz * sBb + (size_t)n0 * ldB;  // B tile row 0

  auto issue = [&](int buf, int kk) {            // one TDM descriptor per wave
    if (wave == 0)
      tdm_load_tile_2d((uint32_t)(uintptr_t)&As[buf][0][0], Ab + kk, K, 64);
    else if (wave == 1)
      tdm_load_tile_2d((uint32_t)(uintptr_t)&Bs[buf][0][0], Bb + kk, ldB, 256);
  };

  v8f c[2][4] = {};

  issue(0, 0);

  const int kIters = K >> 5;
  for (int it = 0; it < kIters; ++it) {
    const int  buf  = it & 1;
    const bool more = (it + 1) < kIters;
    if (more) issue(buf ^ 1, (it + 1) << 5);
    if (wave < 2) {                              // issuing waves gate the DMA
      if (more) __builtin_amdgcn_s_wait_tensorcnt(1);
      else      __builtin_amdgcn_s_wait_tensorcnt(0);
    }
    __syncthreads();                             // tile visible to all waves

    v16bf afr[2], bfr[4];
#pragma unroll
    for (int mi = 0; mi < 2; ++mi) {
      // ISA A-layout (16-bit 16x32): lanes 0-15 take K=0..7 & 16..23, lanes 16-31 K=8..15 & 24..31
      const bf16* p = &As[buf][wm * 32 + mi * 16 + l16][lh * 8];
      v8bf lo = *(const v8bf*)p;
      v8bf hi = *(const v8bf*)(p + 16);
      afr[mi] = __builtin_shufflevector(lo, hi, 0,1,2,3,4,5,6,7,8,9,10,11,12,13,14,15);
    }
#pragma unroll
    for (int ni = 0; ni < 4; ++ni) {
      // B-layout (16-bit 32x16): lane = column, lanes 0-15 take K=0..7 & 16..23, lanes 16-31 K=8..15 & 24..31
      const bf16* p = &Bs[buf][wn * 64 + ni * 16 + l16][lh * 8];
      v8bf lo = *(const v8bf*)p;
      v8bf hi = *(const v8bf*)(p + 16);
      bfr[ni] = __builtin_shufflevector(lo, hi, 0,1,2,3,4,5,6,7,8,9,10,11,12,13,14,15);
    }
#pragma unroll
    for (int mi = 0; mi < 2; ++mi)
#pragma unroll
      for (int ni = 0; ni < 4; ++ni)
        c[mi][ni] = __builtin_amdgcn_wmma_f32_16x16x32_bf16(
            false, afr[mi], false, bfr[ni], (short)0, c[mi][ni], false, false);

    __syncthreads();                             // protect buf^1 before next issue
  }

  // epilogue: C layout VGPR r -> (M = base + r + lh*8, N = base + l16)
#pragma unroll
  for (int mi = 0; mi < 2; ++mi) {
#pragma unroll
    for (int ni = 0; ni < 4; ++ni) {
      const int   col   = n0 + wn * 64 + ni * 16 + l16;
      const int   mbase = m0 + wm * 32 + mi * 16 + lh * 8;
      const float bval  = bias ? bias[col] : 0.0f;
      if (TRANS_OUT) {
        // 8 rows are contiguous along M in [N,ldC] layout -> one 16B store
        v8bf pk;
#pragma unroll
        for (int r = 0; r < 8; ++r)
          pk[r] = (bf16)((c[mi][ni][r] + bval) * scale);
        *(v8bf*)&((bf16*)C)[(size_t)bz * sCb + (size_t)col * ldC + mbase] = pk;
      } else {
#pragma unroll
        for (int r = 0; r < 8; ++r) {
          const float v = (c[mi][ni][r] + bval) * scale;
          const int   m = mbase + r;
          if (OUT_F32)
            ((float*)C)[(size_t)bz * sCb + (size_t)m * ldC + col] = v;
          else
            ((bf16*)C)[(size_t)bz * sCb + (size_t)m * ldC + col] = (bf16)v;
        }
      }
    }
  }
}

// ---------------- row softmax on bf16, in place (cols == 2048) ------------
__global__ __launch_bounds__(256)
void softmax_rows(bf16* __restrict__ P, int cols) {
  bf16* row = P + (size_t)blockIdx.x * cols;
  const int tid  = threadIdx.x;
  const int lane = tid & 31;
  const int wave = tid >> 5;
  __shared__ float red[8];

  const v8bf x = *(const v8bf*)(row + tid * 8);  // one b128 load / thread
  float v[8];
  float mx = -3.4e38f;
#pragma unroll
  for (int j = 0; j < 8; ++j) { v[j] = (float)x[j]; mx = fmaxf(mx, v[j]); }
#pragma unroll
  for (int o = 16; o > 0; o >>= 1) mx = fmaxf(mx, __shfl_xor(mx, o, 32));
  if (lane == 0) red[wave] = mx;
  __syncthreads();
  float m2 = red[0];
#pragma unroll
  for (int w = 1; w < 8; ++w) m2 = fmaxf(m2, red[w]);
  __syncthreads();

  float s = 0.f;
#pragma unroll
  for (int j = 0; j < 8; ++j) { v[j] = __expf(v[j] - m2); s += v[j]; }
#pragma unroll
  for (int o = 16; o > 0; o >>= 1) s += __shfl_xor(s, o, 32);
  if (lane == 0) red[wave] = s;
  __syncthreads();
  float st = 0.f;
#pragma unroll
  for (int w = 0; w < 8; ++w) st += red[w];
  const float inv = 1.0f / st;

  v8bf o;
#pragma unroll
  for (int j = 0; j < 8; ++j) o[j] = (bf16)(v[j] * inv);
  *(v8bf*)(row + tid * 8) = o;                   // one b128 store / thread
}

// ---------------- orchestration -------------------------------------------
extern "C" void kernel_launch(void* const* d_in, const int* in_sizes, int n_in,
                              void* d_out, int out_size, void* d_ws, size_t ws_size,
                              hipStream_t stream) {
  constexpr int    B  = 8, S = 2048, D = 1024;
  constexpr size_t ME = (size_t)B * S;                    // 16384 rows total

  const float* image = (const float*)d_in[0];
  const float* text  = (const float*)d_in[1];
  // d_in[2] = edge_index: only selects the attention branch, unused here
  const float* Wq = (const float*)d_in[3];
  const float* bq = (const float*)d_in[4];
  const float* Wk = (const float*)d_in[5];
  const float* bk = (const float*)d_in[6];
  const float* Wv = (const float*)d_in[7];
  const float* bv = (const float*)d_in[8];

  bf16* ws  = (bf16*)d_ws;
  bf16* Xq  = ws;                      // [ME,D]  bf16 image
  bf16* Xkv = Xq  + ME * D;            // [ME,D]  bf16 text
  bf16* Wqb = Xkv + ME * D;            // [D,D]
  bf16* Wkb = Wqb + (size_t)D * D;
  bf16* Wvb = Wkb + (size_t)D * D;
  bf16* Qb  = Wvb + (size_t)D * D;     // [ME,D]
  bf16* Kb  = Qb  + ME * D;            // [ME,D]
  bf16* Vt  = Kb  + ME * D;            // [D,ME]  (V transposed)
  bf16* Pb  = Xq;                      // scores/probs [B,S,S] alias dead Xq+Xkv
                                       // (B*S*S == 2*ME*D exactly)

  // 1) f32 -> bf16 conversions
  cvt_f32_to_bf16<<<(unsigned)(ME * D / 4 / 256), 256, 0, stream>>>(image, Xq,  (int)(ME * D / 4));
  cvt_f32_to_bf16<<<(unsigned)(ME * D / 4 / 256), 256, 0, stream>>>(text,  Xkv, (int)(ME * D / 4));
  cvt_f32_to_bf16<<<(unsigned)((size_t)D * D / 4 / 256), 256, 0, stream>>>(Wq, Wqb, D * D / 4);
  cvt_f32_to_bf16<<<(unsigned)((size_t)D * D / 4 / 256), 256, 0, stream>>>(Wk, Wkb, D * D / 4);
  cvt_f32_to_bf16<<<(unsigned)((size_t)D * D / 4 / 256), 256, 0, stream>>>(Wv, Wvb, D * D / 4);

  const dim3 blk(256);
  // 2) projections (W is [out,in]=[N,K] row-major -> used directly as Bt)
  gemm_bf16_wmma<false, false><<<dim3(D / 256, ME / 64, 1), blk, 0, stream>>>(
      Xq, 0, Wqb, 0, D, bq, Qb, 0, D, (int)ME, D, D, 1.0f);
  gemm_bf16_wmma<false, false><<<dim3(D / 256, ME / 64, 1), blk, 0, stream>>>(
      Xkv, 0, Wkb, 0, D, bk, Kb, 0, D, (int)ME, D, D, 1.0f);
  gemm_bf16_wmma<true, false><<<dim3(D / 256, ME / 64, 1), blk, 0, stream>>>(
      Xkv, 0, Wvb, 0, D, bv, Vt, 0, (int)ME, (int)ME, D, D, 1.0f);

  // 3) scores = (Q @ K^T) * D^-0.5  (batched; K rows are [S,D] -> Bt directly)
  gemm_bf16_wmma<false, false><<<dim3(S / 256, S / 64, B), blk, 0, stream>>>(
      Qb, (int64_t)S * D, Kb, (int64_t)S * D, D, nullptr,
      Pb, (int64_t)S * S, S, S, S, D, 0.03125f);

  // 4) softmax rows (B*S rows of S)
  softmax_rows<<<(unsigned)(B * S), 256, 0, stream>>>(Pb, S);

  // 5) attended = P @ V  (Bt = V^T rows: elem (d, b*S+s); batch offset b*S, ld = ME)
  gemm_bf16_wmma<false, true><<<dim3(D / 256, S / 64, B), blk, 0, stream>>>(
      Pb, (int64_t)S * S, Vt, (int64_t)S, (int)ME, nullptr,
      d_out, (int64_t)S * D, D, S, D, S, 1.0f);
}
